// NetMHCpan_10_BiLSTM_584115552897
// MI455X (gfx1250) — matrix operand order repaired and checked by reading, not compiled, not run on hardware
//
#include <hip/hip_runtime.h>
#include <hip/hip_bf16.h>

// ---------------------------------------------------------------------------
// NetMHCpan BiLSTM (B=1) for MI455X / gfx1250.
//
// The recurrence is L2-bandwidth bound (all weights ~130MB fit in the 192MB
// L2). whh[4096,1024] is pre-converted to bf16 in the exact per-lane WMMA-B
// layout; each timestep's GEMV runs as v_wmma_f32_16x16x32_bf16 with h
// broadcast into all 16 A rows (same bytes streamed as an ideal GEMV).
// A per-tile-row integer offset is laundered through a VGPR asm so LLVM
// cannot hoist the (timestep-invariant) B-tile loads out of the recurrence
// loop and spill them to scratch, while keeping the loads on the global
// (addrspace-1) path.
// ---------------------------------------------------------------------------

constexpr int kH   = 1024;          // lstm hidden
constexpr int k4H  = 4096;          // 4*H gates / MLP hidden
constexpr int kLP  = 15;            // peptide length
constexpr int kLH  = 372;           // HLA length
constexpr int kIN  = 21;            // embedding dim

// xg layout (floats): [pep_f | pep_b | hla_f | hla_b], each [T][4H]
constexpr size_t kOffPF = 0;
constexpr size_t kOffPB = (size_t)kLP * k4H;
constexpr size_t kOffHF = 2 * kOffPB;
constexpr size_t kOffHB = kOffHF + (size_t)kLH * k4H;
constexpr size_t kXgTotal = kOffHB + (size_t)kLH * k4H;          // 3,170,304 f
constexpr size_t kPackPerDir = (size_t)k4H * kH;                  // 4,194,304 halves

typedef __attribute__((ext_vector_type(16))) __bf16 v16bf;
typedef __attribute__((ext_vector_type(8)))  float  v8f;

__device__ __forceinline__ unsigned short f2bf(float x) {
  union { float f; unsigned u; } v; v.f = x;
  unsigned r = v.u + 0x7FFFu + ((v.u >> 16) & 1u);   // round-to-nearest-even
  return (unsigned short)(r >> 16);
}
__device__ __forceinline__ float sigmoidf(float x) { return 1.0f / (1.0f + expf(-x)); }

// ---------------------------------------------------------------------------
// 1) Pack whh (fp32, row-major [4096,1024]) -> bf16 WMMA-B tiles.
// Tile (jt,kt) covers outputs j in [jt*16,..+16), k in [kt*32,..+32).
// Per-lane layout (wave32, 16 halves = 32B contiguous per lane):
//   lanes 0-15 : column N=lane,     halves e=0..15 -> K = kt*32 + e
//   lanes 16-31: column N=lane-16,  halves e=0..15 -> K = kt*32 + 16 + e
// ---------------------------------------------------------------------------
__global__ void k_pack_whh(const float* __restrict__ w0, const float* __restrict__ w1,
                           const float* __restrict__ w2, const float* __restrict__ w3,
                           unsigned short* __restrict__ packed) {
  size_t idx = (size_t)blockIdx.x * blockDim.x + threadIdx.x;
  if (idx >= 4 * kPackPerDir) return;
  int dir = (int)(idx / kPackPerDir);
  const float* src = (dir == 0) ? w0 : (dir == 1) ? w1 : (dir == 2) ? w2 : w3;
  size_t r  = idx % kPackPerDir;
  int tile  = (int)(r >> 9);          // /512
  int within = (int)(r & 511);
  int lane = within >> 4;
  int e    = within & 15;
  int jt = tile >> 5;                 // 0..255
  int kt = tile & 31;                 // 0..31
  int j = jt * 16 + (lane & 15);
  int k = kt * 32 + ((lane >> 4) << 4) + e;
  packed[idx] = f2bf(src[(size_t)j * kH + k]);
}

// ---------------------------------------------------------------------------
// 2) Input projections: xg[dir][t][g] = x[t]·wih[g,:] + bih[g] + bhh[g]
// ---------------------------------------------------------------------------
__global__ void k_xg(const float* __restrict__ xp, const float* __restrict__ xh,
                     const float* wih_pf, const float* bih_pf, const float* bhh_pf,
                     const float* wih_pb, const float* bih_pb, const float* bhh_pb,
                     const float* wih_hf, const float* bih_hf, const float* bhh_hf,
                     const float* wih_hb, const float* bih_hb, const float* bhh_hb,
                     float* __restrict__ xg) {
  size_t idx = (size_t)blockIdx.x * blockDim.x + threadIdx.x;
  if (idx >= kXgTotal) return;
  const float *x, *wih, *bih, *bhh; size_t rel;
  if (idx < kOffPB)      { rel = idx;          x = xp; wih = wih_pf; bih = bih_pf; bhh = bhh_pf; }
  else if (idx < kOffHF) { rel = idx - kOffPB; x = xp; wih = wih_pb; bih = bih_pb; bhh = bhh_pb; }
  else if (idx < kOffHB) { rel = idx - kOffHF; x = xh; wih = wih_hf; bih = bih_hf; bhh = bhh_hf; }
  else                   { rel = idx - kOffHB; x = xh; wih = wih_hb; bih = bih_hb; bhh = bhh_hb; }
  int t = (int)(rel / k4H);
  int g = (int)(rel % k4H);
  float acc = bih[g] + bhh[g];
  const float* xr = x + (size_t)t * kIN;
  const float* wr = wih + (size_t)g * kIN;
#pragma unroll
  for (int i = 0; i < kIN; ++i) acc += xr[i] * wr[i];
  xg[idx] = acc;
}

// ---------------------------------------------------------------------------
// 3) One block per LSTM direction (0=pep_f,1=pep_b,2=hla_f,3=hla_b).
// 1024 threads = 32 waves; wave w owns output tiles jt = w*8..w*8+7.
// Per step: yv[4096] = whh @ h via WMMA (A = h broadcast to 16 rows, bf16),
// then thread n updates cell n (i,f,g,o gates).  A operand per-lane layout
// (16-bit A 16x32, contiguous to match packed B): lanes 0-15 halves -> K {0..15},
// lanes 16-31 -> K {16..31}; all rows identical so row 0 of D is y.
// ---------------------------------------------------------------------------
__global__ __launch_bounds__(1024, 1)
void k_bilstm(const unsigned short* __restrict__ packed,
              const float* __restrict__ xg,
              float* __restrict__ hfinal) {
  __shared__ float yv[k4H];
  __shared__ __align__(16) unsigned short hbf[kH];

  const int dir = blockIdx.x;
  const int T   = (dir < 2) ? kLP : kLH;
  const bool rev = (dir & 1);
  const size_t xgBase = (dir == 0) ? kOffPF : (dir == 1) ? kOffPB
                       : (dir == 2) ? kOffHF : kOffHB;
  const unsigned short* wp = packed + (size_t)dir * kPackPerDir;

  const int tid  = threadIdx.x;          // == cell index n (0..1023)
  const int lane = tid & 31;
  const int wave = tid >> 5;
  const int aoff = (lane >> 4) << 4;     // 0 for lanes 0-15, 16 for 16-31
  const int laneOff = lane * 16;         // halves

  float c = 0.0f, h = 0.0f;
  hbf[tid] = 0;                          // h0 = 0
  __syncthreads();

  union AU { uint4 q[2]; v16bf v; };

  for (int s = 0; s < T; ++s) {
    // ---------- phase A: yv = whh @ h (WMMA, bf16) ----------
    for (int jl = 0; jl < 8; ++jl) {
      const int jt = wave * 8 + jl;
      // Opaque base offset (halves) of this tile-row's 32 B-tiles.
      // Laundered through a VGPR asm ("+v": legal for divergent values) so
      // LLVM cannot prove the loads timestep-invariant (which previously made
      // it hoist 8KB/lane of tiles out of the s-loop and spill to scratch).
      // Laundering the offset, not the pointer, keeps addrspace(1) inference
      // on wp so these remain global_load_b128.
      unsigned int tbase = (unsigned int)(jt * 32 * 512) + (unsigned int)laneOff;
      asm volatile("" : "+v"(tbase));
      v8f acc0 = {}; v8f acc1 = {};      // two chains break D->C serialization
#pragma unroll 4
      for (int kt = 0; kt < 32; kt += 2) {
        AU a0, a1, b0, b1;
        const int k0 = kt * 32, k1 = k0 + 32;
        a0.q[0] = *(const uint4*)&hbf[k0 + aoff];
        a0.q[1] = *(const uint4*)&hbf[k0 + 8 + aoff];
        a1.q[0] = *(const uint4*)&hbf[k1 + aoff];
        a1.q[1] = *(const uint4*)&hbf[k1 + 8 + aoff];
        const uint4* p0 = (const uint4*)(wp + (size_t)tbase + (size_t)kt * 512);
        const uint4* p1 = (const uint4*)(wp + (size_t)tbase + (size_t)(kt + 1) * 512);
        if (kt + 4 < 32)
          __builtin_prefetch((const void*)(wp + (size_t)tbase + (size_t)(kt + 4) * 512), 0, 3);
        b0.q[0] = p0[0]; b0.q[1] = p0[1];
        b1.q[0] = p1[0]; b1.q[1] = p1[1];
        acc0 = __builtin_amdgcn_wmma_f32_16x16x32_bf16(false, a0.v, false, b0.v,
                                                       (short)0, acc0, false, false);
        acc1 = __builtin_amdgcn_wmma_f32_16x16x32_bf16(false, a1.v, false, b1.v,
                                                       (short)0, acc1, false, false);
      }
      v8f y = acc0 + acc1;
      if (lane < 16) yv[jt * 16 + lane] = y[0];   // row M=0 lives in VGPR0, lanes 0-15
    }
    __syncthreads();

    // ---------- phase B: pointwise LSTM cell update ----------
    const int t = rev ? (T - 1 - s) : s;
    const float* xgt = xg + xgBase + (size_t)t * k4H;
    float gi = xgt[tid]            + yv[tid];
    float gf = xgt[kH + tid]       + yv[kH + tid];
    float gg = xgt[2 * kH + tid]   + yv[2 * kH + tid];
    float go = xgt[3 * kH + tid]   + yv[3 * kH + tid];
    c = sigmoidf(gf) * c + sigmoidf(gi) * tanhf(gg);
    h = sigmoidf(go) * tanhf(c);
    hbf[tid] = f2bf(h);
    __syncthreads();
  }
  hfinal[(size_t)dir * kH + tid] = h;    // order pep_f|pep_b|hla_f|hla_b == concat
}

// ---------------------------------------------------------------------------
// 4) MLP: h1 = relu(w1 @ xcat + b1)   (4096x4096 GEMV, one-shot, fp32)
// ---------------------------------------------------------------------------
__global__ void k_mlp1(const float* __restrict__ w1, const float* __restrict__ b1,
                       const float* __restrict__ xcat, float* __restrict__ h1) {
  int j = blockIdx.x * blockDim.x + threadIdx.x;
  if (j >= k4H) return;
  float acc = b1[j];
  const float* wr = w1 + (size_t)j * k4H;
  for (int k = 0; k < k4H; k += 4) {
    float4 w = *(const float4*)(wr + k);
    float4 x = *(const float4*)(xcat + k);
    acc += w.x * x.x + w.y * x.y + w.z * x.z + w.w * x.w;
  }
  h1[j] = fmaxf(acc, 0.0f);
}

// 5) out = w2 · h1 + b2
__global__ void k_mlp2(const float* __restrict__ w2, const float* __restrict__ b2,
                       const float* __restrict__ h1, float* __restrict__ out) {
  __shared__ float red[1024];
  int tid = threadIdx.x;
  float s = 0.0f;
  for (int j = tid; j < k4H; j += 1024) s += w2[j] * h1[j];
  red[tid] = s;
  __syncthreads();
  for (int off = 512; off > 0; off >>= 1) {
    if (tid < off) red[tid] += red[tid + off];
    __syncthreads();
  }
  if (tid == 0) out[0] = red[0] + b2[0];
}

// ---------------------------------------------------------------------------
extern "C" void kernel_launch(void* const* d_in, const int* in_sizes, int n_in,
                              void* d_out, int out_size, void* d_ws, size_t ws_size,
                              hipStream_t stream) {
  const float* input_pep = (const float*)d_in[0];
  const float* input_hla = (const float*)d_in[1];
  const float* pep_wih_f = (const float*)d_in[2];
  const float* pep_whh_f = (const float*)d_in[3];
  const float* pep_bih_f = (const float*)d_in[4];
  const float* pep_bhh_f = (const float*)d_in[5];
  const float* pep_wih_b = (const float*)d_in[6];
  const float* pep_whh_b = (const float*)d_in[7];
  const float* pep_bih_b = (const float*)d_in[8];
  const float* pep_bhh_b = (const float*)d_in[9];
  const float* hla_wih_f = (const float*)d_in[10];
  const float* hla_whh_f = (const float*)d_in[11];
  const float* hla_bih_f = (const float*)d_in[12];
  const float* hla_bhh_f = (const float*)d_in[13];
  const float* hla_wih_b = (const float*)d_in[14];
  const float* hla_whh_b = (const float*)d_in[15];
  const float* hla_bih_b = (const float*)d_in[16];
  const float* hla_bhh_b = (const float*)d_in[17];
  const float* w1 = (const float*)d_in[18];
  const float* b1 = (const float*)d_in[19];
  const float* w2 = (const float*)d_in[20];
  const float* b2 = (const float*)d_in[21];
  (void)in_sizes; (void)n_in; (void)out_size; (void)ws_size;

  // workspace layout (bytes)
  char* ws = (char*)d_ws;
  const size_t packBytes = 4 * kPackPerDir * sizeof(unsigned short); // 33,554,432
  const size_t xgBytes   = kXgTotal * sizeof(float);                 // 12,681,216
  unsigned short* packed = (unsigned short*)ws;
  float* xg     = (float*)(ws + packBytes);
  float* hfinal = (float*)(ws + packBytes + xgBytes);
  float* h1     = (float*)(ws + packBytes + xgBytes + 4 * kH * sizeof(float));

  {
    size_t total = 4 * kPackPerDir;
    k_pack_whh<<<(unsigned)((total + 255) / 256), 256, 0, stream>>>(
        pep_whh_f, pep_whh_b, hla_whh_f, hla_whh_b, packed);
  }
  {
    size_t total = kXgTotal;
    k_xg<<<(unsigned)((total + 255) / 256), 256, 0, stream>>>(
        input_pep, input_hla,
        pep_wih_f, pep_bih_f, pep_bhh_f,
        pep_wih_b, pep_bih_b, pep_bhh_b,
        hla_wih_f, hla_bih_f, hla_bhh_f,
        hla_wih_b, hla_bih_b, hla_bhh_b, xg);
  }
  k_bilstm<<<4, 1024, 0, stream>>>(packed, xg, hfinal);
  k_mlp1<<<k4H / 256, 256, 0, stream>>>(w1, b1, hfinal, h1);
  k_mlp2<<<1, 1024, 0, stream>>>(w2, b2, h1, (float*)d_out);
}
